// GCN_61125974556852
// MI455X (gfx1250) — matrix operand order repaired and checked by reading, not compiled, not run on hardware
//
#include <hip/hip_runtime.h>
#include <hip/hip_bf16.h>

typedef __attribute__((ext_vector_type(16))) __bf16 v16bf;
typedef __attribute__((ext_vector_type(8)))  float  v8f;

#define IN_F   4096
#define HID_F  512
#define NCLS   10
#define KTILES (IN_F / 32)

// ---------------- helpers ----------------
__device__ __forceinline__ unsigned short f2bf(float f) {
    union { float f; unsigned u; } v; v.f = f;
    unsigned r = v.u + 0x7FFFu + ((v.u >> 16) & 1u);   // round-to-nearest-even
    return (unsigned short)(r >> 16);
}
__device__ __forceinline__ unsigned pack_bf2(float a, float b) {
    union { float f; unsigned u; } x, y; x.f = a; y.f = b;
    unsigned ra = x.u + 0x7FFFu + ((x.u >> 16) & 1u);
    unsigned rb = y.u + 0x7FFFu + ((y.u >> 16) & 1u);
    return (ra >> 16) | (rb & 0xFFFF0000u);
}

// ---------------- degree / norm ----------------
__global__ void deg_init_kernel(float* deg, int N) {
    int i = blockIdx.x * blockDim.x + threadIdx.x;
    if (i < N) deg[i] = 1.0f;                      // self loop
}
__global__ void deg_count_kernel(const long long* ei, float* deg, int E) {
    int e = blockIdx.x * blockDim.x + threadIdx.x;
    if (e < E) {
        int d = (int)ei[(size_t)E + e];            // dst row of edge_index
        atomicAdd(&deg[d], 1.0f);
    }
}
__global__ void dinv_kernel(float* deg, int N) {
    int i = blockIdx.x * blockDim.x + threadIdx.x;
    if (i < N) {
        float d = deg[i];
        deg[i] = d > 0.f ? rsqrtf(d) : 0.f;        // in place -> dinv
    }
}

// ---------------- W1 -> W1^T (bf16) ----------------
__global__ void w1_transpose_kernel(const float* __restrict__ W1,
                                    unsigned short* __restrict__ W1T) {
    __shared__ float tile[32][33];
    int nb = blockIdx.x * 32;            // output-feature block (512 total)
    int kb = blockIdx.y * 32;            // input-feature block (4096 total)
    int tx = threadIdx.x, ty = threadIdx.y;   // 32 x 8
    #pragma unroll
    for (int i = 0; i < 32; i += 8)
        tile[ty + i][tx] = W1[(size_t)(kb + ty + i) * HID_F + nb + tx];
    __syncthreads();
    #pragma unroll
    for (int i = 0; i < 32; i += 8)
        W1T[(size_t)(nb + ty + i) * IN_F + kb + tx] = f2bf(tile[tx][ty + i]);
}

// ---------------- GEMM1: H1 = x @ W1  (bf16 WMMA, fp32 acc) ----------------
// block: 256 threads (8 waves), tile 128(M) x 256(N) x 32(K), double-buffered LDS.
// BLOCK_N=256 halves DRAM re-reads of x (327 MB fp32, doesn't fit in 192 MB L2).
#define LDS_STRIDE 40   // 32 bf16 + 8 pad
__global__ void __launch_bounds__(256)
gemm1_wmma_kernel(const float* __restrict__ x,
                  const unsigned short* __restrict__ W1T,
                  float* __restrict__ H1, int Nrows) {
    __shared__ __align__(16) unsigned short As[2][128 * LDS_STRIDE];
    __shared__ __align__(16) unsigned short Bs[2][256 * LDS_STRIDE];

    const int tid    = threadIdx.x;
    const int lane   = tid & 31;
    const int wave   = tid >> 5;          // 0..7
    const int wave_m = wave & 1;          // 2 waves along M (64 rows each)
    const int wave_n = wave >> 1;         // 4 waves along N (64 cols each)
    const int lh     = lane >> 4;         // half-wave id
    const int lm     = lane & 15;

    const int m_base = blockIdx.y * 128;
    const int n_base = blockIdx.x * 256;

    const int ld_row  = tid >> 1;         // 0..127 (A tile)
    const int ld_half = tid & 1;          // 16-element halves of a 32-elem row

    // Clamp A row: rows beyond Nrows are never stored, so their tile contents
    // are don't-care — clamping keeps loads in-bounds and branch-free.
    const int arow = min(m_base + ld_row, Nrows - 1);
    const float*          aG = x   + (size_t)arow * IN_F + ld_half * 16;
    const unsigned short* bG = W1T + (size_t)(n_base + tid) * IN_F;   // 1 row/thread

    v8f acc[4][4];
    #pragma unroll
    for (int mi = 0; mi < 4; ++mi)
        #pragma unroll
        for (int ni = 0; ni < 4; ++ni)
            acc[mi][ni] = (v8f)0.0f;

    union Frag { uint4 q[2]; v16bf v; };

    auto load_tiles = [&](int k0, int buf) {
        // ---- A tile: fp32 -> bf16 via VALU, ds_store ----
        const float4* src = (const float4*)(aG + k0);
        unsigned* dst = (unsigned*)&As[buf][ld_row * LDS_STRIDE + ld_half * 16];
        #pragma unroll
        for (int j = 0; j < 4; ++j) {
            float4 f = src[j];
            dst[2 * j + 0] = pack_bf2(f.x, f.y);
            dst[2 * j + 1] = pack_bf2(f.z, f.w);
        }
        // ---- B tile: pure bf16 copy -> CDNA5 async global->LDS (ASYNCcnt) ----
        // one row (32 bf16 = 64 B) per thread; INST_OFFSET applies to both sides
        unsigned ldsOff = (unsigned)(size_t)&Bs[buf][tid * LDS_STRIDE];
        const unsigned short* g0 = bG + k0;
        asm volatile("global_load_async_to_lds_b128 %0, %1, off"
                     :: "v"(ldsOff), "v"(g0) : "memory");
        asm volatile("global_load_async_to_lds_b128 %0, %1, off offset:16"
                     :: "v"(ldsOff), "v"(g0) : "memory");
        asm volatile("global_load_async_to_lds_b128 %0, %1, off offset:32"
                     :: "v"(ldsOff), "v"(g0) : "memory");
        asm volatile("global_load_async_to_lds_b128 %0, %1, off offset:48"
                     :: "v"(ldsOff), "v"(g0) : "memory");
        // prefetch A two K-steps ahead
        if (k0 + 64 < IN_F) __builtin_prefetch(aG + k0 + 64, 0, 1);
    };

    auto compute = [&](int buf) {
        // load ALL fragments first -> one dscnt wait, then a dense WMMA burst
        Frag bfr[4];
        #pragma unroll
        for (int ni = 0; ni < 4; ++ni) {
            int col = wave_n * 64 + ni * 16 + lm;
            const unsigned short* bp = &Bs[buf][col * LDS_STRIDE + lh * 16];
            bfr[ni].q[0] = ((const uint4*)bp)[0];
            bfr[ni].q[1] = ((const uint4*)bp)[1];
        }
        Frag af[4];
        #pragma unroll
        for (int mi = 0; mi < 4; ++mi) {
            int row = wave_m * 64 + mi * 16 + lm;
            const unsigned short* ap = &As[buf][row * LDS_STRIDE];
            // lanes 0-15: K 0-15 ; lanes 16-31: K 16-31
            af[mi].q[0] = *(const uint4*)(ap + (lh ? 16 : 0));
            af[mi].q[1] = *(const uint4*)(ap + (lh ? 24 : 8));
        }
        #pragma unroll
        for (int mi = 0; mi < 4; ++mi)
            #pragma unroll
            for (int ni = 0; ni < 4; ++ni)
                acc[mi][ni] = __builtin_amdgcn_wmma_f32_16x16x32_bf16(
                    false, af[mi].v, false, bfr[ni].v,
                    (short)0, acc[mi][ni], false, false);
    };

    // prologue: fill buffer 0
    load_tiles(0, 0);
    asm volatile("s_wait_asynccnt 0" ::: "memory");
    __syncthreads();

    for (int kt = 0; kt < KTILES; ++kt) {
        const int cur = kt & 1;
        if (kt + 1 < KTILES) load_tiles((kt + 1) * 32, cur ^ 1);
        compute(cur);
        asm volatile("s_wait_asynccnt 0" ::: "memory");
        __syncthreads();
    }

    // ---- store C: VGPR v -> row v (lanes 0-15) / v+8 (lanes 16-31), col = lane&15
    #pragma unroll
    for (int mi = 0; mi < 4; ++mi) {
        #pragma unroll
        for (int ni = 0; ni < 4; ++ni) {
            int col = n_base + wave_n * 64 + ni * 16 + lm;
            #pragma unroll
            for (int v = 0; v < 8; ++v) {
                int row = m_base + wave_m * 64 + mi * 16 + lh * 8 + v;
                if (row < Nrows) H1[(size_t)row * HID_F + col] = acc[mi][ni][v];
            }
        }
    }
}

// ---------------- propagation layer 1 ----------------
__global__ void agg1_init_kernel(const float* __restrict__ H1,
                                 const float* __restrict__ dinv,
                                 const float* __restrict__ b1,
                                 float* __restrict__ out1, int N) {
    int idx = blockIdx.x * blockDim.x + threadIdx.x;
    if (idx < N * HID_F) {
        int i = idx / HID_F, f = idx - i * HID_F;
        float di = dinv[i];
        out1[idx] = H1[idx] * di * di + b1[f];   // self-loop + bias
    }
}
__global__ void agg1_edge_kernel(const long long* __restrict__ ei,
                                 const float* __restrict__ H1,
                                 const float* __restrict__ dinv,
                                 float* __restrict__ out1, int E) {
    int e = blockIdx.x;                       // one block per edge
    int s = (int)ei[e];
    int d = (int)ei[(size_t)E + e];
    float norm = dinv[s] * dinv[d];
    int f = threadIdx.x * 4;                  // 128 threads * 4 = 512
    float4 h = *(const float4*)(H1 + (size_t)s * HID_F + f);
    float* o = out1 + (size_t)d * HID_F + f;
    atomicAdd(o + 0, h.x * norm);
    atomicAdd(o + 1, h.y * norm);
    atomicAdd(o + 2, h.z * norm);
    atomicAdd(o + 3, h.w * norm);
}

// ---------------- GEMM2: H2 = relu(out1) @ W2  (wave per row) ----------------
__global__ void __launch_bounds__(256)
gemm2_kernel(const float* __restrict__ out1, const float* __restrict__ W2,
             float* __restrict__ H2, int N) {
    __shared__ float W2s[HID_F * NCLS];
    for (int i = threadIdx.x; i < HID_F * NCLS; i += 256) W2s[i] = W2[i];
    __syncthreads();
    int lane = threadIdx.x & 31;
    int row  = blockIdx.x * 8 + (threadIdx.x >> 5);
    if (row >= N) return;
    float acc[NCLS];
    #pragma unroll
    for (int c = 0; c < NCLS; ++c) acc[c] = 0.f;
    const float* hp = out1 + (size_t)row * HID_F;
    #pragma unroll 4
    for (int j = 0; j < 16; ++j) {
        int f = lane + 32 * j;
        float v = fmaxf(hp[f], 0.f);          // fused ReLU
        const float* w = &W2s[f * NCLS];
        #pragma unroll
        for (int c = 0; c < NCLS; ++c) acc[c] = fmaf(v, w[c], acc[c]);
    }
    #pragma unroll
    for (int c = 0; c < NCLS; ++c) {
        float s = acc[c];
        #pragma unroll
        for (int off = 16; off > 0; off >>= 1) s += __shfl_xor(s, off, 32);
        if (lane == 0) H2[(size_t)row * NCLS + c] = s;
    }
}

// ---------------- propagation layer 2 ----------------
__global__ void agg2_init_kernel(const float* __restrict__ H2,
                                 const float* __restrict__ dinv,
                                 const float* __restrict__ b2,
                                 float* __restrict__ out2, int N) {
    int idx = blockIdx.x * blockDim.x + threadIdx.x;
    if (idx < N * NCLS) {
        int i = idx / NCLS, c = idx - i * NCLS;
        float di = dinv[i];
        out2[idx] = H2[idx] * di * di + b2[c];
    }
}
__global__ void agg2_edge_kernel(const long long* __restrict__ ei,
                                 const float* __restrict__ H2,
                                 const float* __restrict__ dinv,
                                 float* __restrict__ out2, int E) {
    int tid = blockIdx.x * blockDim.x + threadIdx.x;
    int e = tid >> 4, c = tid & 15;
    if (e < E && c < NCLS) {
        int s = (int)ei[e];
        int d = (int)ei[(size_t)E + e];
        float norm = dinv[s] * dinv[d];
        atomicAdd(&out2[(size_t)d * NCLS + c], H2[(size_t)s * NCLS + c] * norm);
    }
}

// ---------------- log-softmax ----------------
__global__ void log_softmax_kernel(const float* __restrict__ in,
                                   float* __restrict__ out, int N) {
    int i = blockIdx.x * blockDim.x + threadIdx.x;
    if (i >= N) return;
    const float* r = in + (size_t)i * NCLS;
    float v[NCLS], m = -1e30f;
    #pragma unroll
    for (int c = 0; c < NCLS; ++c) { v[c] = r[c]; m = fmaxf(m, v[c]); }
    float s = 0.f;
    #pragma unroll
    for (int c = 0; c < NCLS; ++c) s += __expf(v[c] - m);
    float lse = m + __logf(s);
    float* o = out + (size_t)i * NCLS;
    #pragma unroll
    for (int c = 0; c < NCLS; ++c) o[c] = v[c] - lse;
}

// ---------------- launcher ----------------
extern "C" void kernel_launch(void* const* d_in, const int* in_sizes, int n_in,
                              void* d_out, int out_size, void* d_ws, size_t ws_size,
                              hipStream_t stream) {
    const float*     x   = (const float*)d_in[0];
    const long long* ei  = (const long long*)d_in[1];
    const float*     W1  = (const float*)d_in[2];
    const float*     b1  = (const float*)d_in[3];
    const float*     W2  = (const float*)d_in[4];
    const float*     b2  = (const float*)d_in[5];
    float*           out = (float*)d_out;

    const int N = in_sizes[0] / IN_F;      // 20000
    const int E = in_sizes[1] / 2;         // 640000

    // workspace carve-out (256-B aligned chunks)
    char* ws = (char*)d_ws;
    size_t off = 0;
    auto carve = [&](size_t bytes) {
        char* p = ws + off;
        off += (bytes + 255) & ~(size_t)255;
        return p;
    };
    float*          dinv = (float*)carve((size_t)N * 4);
    unsigned short* W1T  = (unsigned short*)carve((size_t)HID_F * IN_F * 2);
    float*          H1   = (float*)carve((size_t)N * HID_F * 4);
    float*          out1 = (float*)carve((size_t)N * HID_F * 4);
    float*          H2   = (float*)carve((size_t)N * NCLS * 4);
    float*          out2 = (float*)carve((size_t)N * NCLS * 4);
    (void)ws_size; (void)n_in; (void)out_size;

    // 1) degrees -> dinv
    deg_init_kernel <<<(N + 255) / 256, 256, 0, stream>>>(dinv, N);
    deg_count_kernel<<<(E + 255) / 256, 256, 0, stream>>>(ei, dinv, E);
    dinv_kernel     <<<(N + 255) / 256, 256, 0, stream>>>(dinv, N);

    // 2) W1^T in bf16 (one-time, tiny)
    w1_transpose_kernel<<<dim3(HID_F / 32, IN_F / 32), dim3(32, 8), 0, stream>>>(W1, W1T);

    // 3) big GEMM via bf16 WMMA (128x256 tile, double-buffered, async B copies)
    gemm1_wmma_kernel<<<dim3(HID_F / 256, (N + 127) / 128), 256, 0, stream>>>(x, W1T, H1, N);

    // 4) propagate layer 1 (self-loop+bias init, then edge scatter-add)
    agg1_init_kernel<<<((size_t)N * HID_F + 255) / 256, 256, 0, stream>>>(H1, dinv, b1, out1, N);
    agg1_edge_kernel<<<E, 128, 0, stream>>>(ei, H1, dinv, out1, E);

    // 5) GEMM2 with fused ReLU
    gemm2_kernel<<<(N + 7) / 8, 256, 0, stream>>>(out1, W2, H2, N);

    // 6) propagate layer 2
    agg2_init_kernel<<<((size_t)N * NCLS + 255) / 256, 256, 0, stream>>>(H2, dinv, b2, out2, N);
    agg2_edge_kernel<<<((size_t)E * 16 + 255) / 256, 256, 0, stream>>>(ei, H2, dinv, out2, E);

    // 7) log-softmax -> d_out
    log_softmax_kernel<<<(N + 255) / 256, 256, 0, stream>>>(out2, out, N);
}
